// RNN_31344671326906
// MI455X (gfx1250) — matrix operand order repaired and checked by reading, not compiled, not run on hardware
//
#include <hip/hip_runtime.h>
#include <hip/hip_bf16.h>

typedef __attribute__((ext_vector_type(16))) __bf16 v16bf;
typedef __attribute__((ext_vector_type(8)))  __bf16 v8bf;
typedef __attribute__((ext_vector_type(8)))  float  v8f;

union FragU { v16bf v; v8bf h[2]; };

#define T_STEPS 2048
#define BATCH   32
#define HCN     512
#define ICN     512
#define MROWS   (T_STEPS * BATCH)   // 65536
#define SC_NWG  8

#if defined(__HIP_DEVICE_COMPILE__) && __has_builtin(__builtin_amdgcn_tanhf)
#define TANHF(x) __builtin_amdgcn_tanhf(x)
#else
#define TANHF(x) tanhf(x)
#endif

// ---- CDNA5 async-to-LDS (CDNA5 ISA), GV mode ----
// 3 copies per thread; variant with a *fused* counted wait: the wait is glued
// to the issues inside one asm unit, so "<=3 outstanding" always means
// "everything except the 3 just-issued has landed" (async loads complete in
// order, ISA §4.1) no matter how the scheduler places the block.
struct TileXfer { unsigned la0, la1, lb; const void *ga0, *ga1, *gb; };

__device__ __forceinline__ void issue3(const TileXfer& t) {
  asm volatile(
      "global_load_async_to_lds_b128 %0, %3, off\n\t"
      "global_load_async_to_lds_b128 %1, %4, off\n\t"
      "global_load_async_to_lds_b128 %2, %5, off"
      :: "v"(t.la0), "v"(t.la1), "v"(t.lb), "v"(t.ga0), "v"(t.ga1), "v"(t.gb)
      : "memory");
}
__device__ __forceinline__ void issue3_wait3(const TileXfer& t) {
  asm volatile(
      "global_load_async_to_lds_b128 %0, %3, off\n\t"
      "global_load_async_to_lds_b128 %1, %4, off\n\t"
      "global_load_async_to_lds_b128 %2, %5, off\n\t"
      "s_wait_asynccnt 0x3"
      :: "v"(t.la0), "v"(t.la1), "v"(t.lb), "v"(t.ga0), "v"(t.ga1), "v"(t.gb)
      : "memory");
}
__device__ __forceinline__ void wait_async0() {
#if defined(__HIP_DEVICE_COMPILE__) && __has_builtin(__builtin_amdgcn_s_wait_asynccnt)
  __builtin_amdgcn_s_wait_asynccnt(0);
#else
  asm volatile("s_wait_asynccnt 0x0" ::: "memory");
#endif
}

// ---- WMMA fragment slot maps (CDNA5 ISA 7.12.2, wave32) ----
// 16-bit B 32x16: lane = N + 16*((K>>3)&1); elem = (K&7) + 8*(K>=16)
__device__ __forceinline__ int b_slot(int kk, int nn) {
  return (nn + ((kk & 8) ? 16 : 0)) * 16 + ((kk & 16) ? 8 : 0) + (kk & 7);
}

// ============================================================================
// f32 -> bf16 bulk convert (8 elems/thread).  Used once for x and per-layer Wi;
// halves the GEMM's streaming read traffic on the 23.3 TB/s HBM.
// ============================================================================
__global__ __launch_bounds__(256) void cvt_kernel(const float* __restrict__ src,
                                                  __bf16* __restrict__ dst, long n) {
  long i = ((long)blockIdx.x * 256 + threadIdx.x) * 8;
  if (i + 8 <= n) {
    float4 a = *(const float4*)&src[i];
    float4 b = *(const float4*)&src[i + 4];
    v8bf o;
    o[0] = (__bf16)a.x; o[1] = (__bf16)a.y; o[2] = (__bf16)a.z; o[3] = (__bf16)a.w;
    o[4] = (__bf16)b.x; o[5] = (__bf16)b.y; o[6] = (__bf16)b.z; o[7] = (__bf16)b.w;
    *(v8bf*)&dst[i] = o;
  }
}

// ============================================================================
// Pre-swizzle Wh (one layer) into bf16 B-fragment order so the scan can load
// its 16 fragments straight into registers.  WG wg owns columns [wg*64, +64).
// ============================================================================
__global__ __launch_bounds__(256) void whprep_kernel(const float* __restrict__ Wh,
                                                     __bf16* __restrict__ WhFrag) {
  const int wg = blockIdx.x, tid = threadIdx.x, n0 = wg * 64;
#pragma unroll 4
  for (int j = 0; j < 32; ++j) {
    int f4 = tid + j * 256;                // 8192 float4 total (512 k x 64 n)
    int nl = f4 >> 7;
    int kq = (f4 & 127) * 4;
    float4 v = *(const float4*)&Wh[(long)(n0 + nl) * HCN + kq];
    float vv[4] = {v.x, v.y, v.z, v.w};
#pragma unroll
    for (int c = 0; c < 4; ++c) {
      int k = kq + c;
      WhFrag[((size_t)wg * 64 + (k >> 5) * 4 + (nl >> 4)) * 512 +
             b_slot(k & 31, nl & 15)] = (__bf16)vv[c];
    }
  }
}

// ============================================================================
// Input projection GEMM: P[m,n] = sum_k Xb[m,k]*Wb[n,k] + bias[n]  (bf16 WMMA)
// Tiles 128(M) x 64(N), K-step 32.  Raw bf16 tiles DMA'd into LDS with
// global_load_async_to_lds_b128, double buffered.  Per iteration:
//   [issue next tile + fused s_wait_asynccnt 3]  -> confirms current tile
//   barrier -> ds-load fragments -> 4 WMMAs -> barrier
// so the next tile's DMA overlaps the whole compute phase, and every
// buffer hand-off is barrier-separated.  Rows padded to 80B (bank spread).
// ============================================================================
__global__ __launch_bounds__(256) void xproj_kernel(
    const __bf16* __restrict__ X, const __bf16* __restrict__ W,
    const float* __restrict__ bias, float* __restrict__ P) {
  __shared__ __align__(16) unsigned char ldsA[2][128 * 80];  // 20 KB
  __shared__ __align__(16) unsigned char ldsB[2][64 * 80];   // 10 KB
  const int tid  = threadIdx.x;
  const int lane = tid & 31;
  const int wv   = tid >> 5;
  const int wm   = wv >> 1;               // 0..3 -> 32 rows
  const int wn   = wv & 1;                // 0..1 -> 32 cols
  const long r0  = (long)blockIdx.x * 128;
  const int  n0  = blockIdx.y * 64;

  // per-thread transfer slots (fixed): A gets 2 x 16B, B gets 1 x 16B
  const int ra0 = tid >> 2,        ca0 = tid & 3;          // A rows 0..63
  const int ra1 = (tid + 256) >> 2, ca1 = tid & 3;         // A rows 64..127
  const int nlb = tid >> 2,        cb  = tid & 3;          // B rows 0..63

  auto tile_addrs = [&](int kt, int buf) {
    const int k0 = kt * 32;
    TileXfer t;
    t.la0 = (unsigned)(size_t)&ldsA[buf][ra0 * 80 + ca0 * 16];
    t.ga0 = &X[(r0 + ra0) * ICN + k0 + ca0 * 8];
    t.la1 = (unsigned)(size_t)&ldsA[buf][ra1 * 80 + ca1 * 16];
    t.ga1 = &X[(r0 + ra1) * ICN + k0 + ca1 * 8];
    t.lb  = (unsigned)(size_t)&ldsB[buf][nlb * 80 + cb * 16];
    t.gb  = &W[(size_t)(n0 + nlb) * ICN + k0 + cb * 8];
    return t;
  };

  v8f acc[2][2] = {};
  issue3(tile_addrs(0, 0));
  const int rrow = lane & 15, half = lane >> 4;

  for (int kt = 0; kt < 16; ++kt) {
    const int cur = kt & 1;
    if (kt + 1 < 16) {
      issue3_wait3(tile_addrs(kt + 1, 1 - cur));  // launch nxt, confirm cur
    } else {
      wait_async0();
    }
    __syncthreads();        // everyone's copies for buf[cur] published
    FragU ua[2], ub[2];
#pragma unroll
    for (int i = 0; i < 2; ++i) {
      const unsigned char* ar = &ldsA[cur][((wm * 2 + i) * 16 + rrow) * 80];
      ua[i].h[0] = *(const v8bf*)(ar + half * 16);        // K = half*8 .. +7
      ua[i].h[1] = *(const v8bf*)(ar + 32 + half * 16);   // K = 16+half*8 .. +7
      const unsigned char* br = &ldsB[cur][((wn * 2 + i) * 16 + rrow) * 80];
      ub[i].h[0] = *(const v8bf*)(br + half * 16);        // K = half*8 .. +7
      ub[i].h[1] = *(const v8bf*)(br + 32 + half * 16);   // K = 16+half*8 .. +7
    }
#pragma unroll
    for (int i = 0; i < 2; ++i)
#pragma unroll
      for (int jj = 0; jj < 2; ++jj)
        acc[i][jj] = __builtin_amdgcn_wmma_f32_16x16x32_bf16(
            false, ua[i].v, false, ub[jj].v, (short)0, acc[i][jj], false, false);
    __syncthreads();        // all waves done reading buf[cur] -> safe to refill
  }
  // epilogue: D layout M = e + 8*(lane>=16), N = lane&15
  const int Nin = lane & 15, mhalf = (lane >> 4) * 8;
#pragma unroll
  for (int i = 0; i < 2; ++i)
#pragma unroll
    for (int jj = 0; jj < 2; ++jj) {
      int gcol = n0 + wn * 32 + jj * 16 + Nin;
      float bb = bias[gcol];
#pragma unroll
      for (int e = 0; e < 8; ++e) {
        long grow = r0 + wm * 32 + i * 16 + mhalf + e;
        P[grow * HCN + gcol] = acc[i][jj][e] + bb;
      }
    }
}

// ============================================================================
// Recurrent scan: h_{t+1} = tanh(P[t] + h_t @ Wh^T + bh)
// 8 persistent WGs, zero LDS.  Wh fragments live in registers for all 2048
// steps; h ping-pongs in global memory *already in bf16 A-fragment order*
// (written by the epilogue), so each step is 32 global_load_b128 -> 16 WMMAs.
// Agent-scope atomic barrier between steps; P rows prefetched during compute.
// ============================================================================
__global__ __launch_bounds__(256) void scan_kernel(
    const float* __restrict__ P, const __bf16* __restrict__ WhFrag,
    const float* __restrict__ bh, float* __restrict__ ys,
    __bf16* __restrict__ ysb /*optional bf16 mirror*/, float* __restrict__ hs,
    __bf16* __restrict__ hfrag, unsigned* __restrict__ cnt, int Tn) {
  const int tid  = threadIdx.x;
  const int lane = tid & 31;
  const int w    = tid >> 5;
  const int mt   = w >> 2;   // 0..1 (batch tile)
  const int nt   = w & 3;    // 0..3 (column tile)
  const int n0   = blockIdx.x * 64;

  // hold this wave's 16 Wh B-fragments (K=512) in registers for the whole scan
  v16bf wb[16];
#pragma unroll
  for (int kt = 0; kt < 16; ++kt)
    wb[kt] = *(const v16bf*)&WhFrag[((size_t)blockIdx.x * 64 + kt * 4 + nt) * 512 + lane * 16];

  const int Nin   = lane & 15;
  const int mhalf = (lane >> 4) * 8;
  const int hc    = n0 + nt * 16 + Nin;
  const float bias = bh[hc];
  // precomputed slot pieces for writing h back in A-fragment order
  const int fragbase = ((hc >> 5) * 2 + mt) * 512;
  const int lane_hi  = ((hc >> 3) & 1) << 4;
  const int e_a      = ((hc & 16) ? 8 : 0) + (hc & 7);

  for (int t = 0; t < Tn; ++t) {
    const __bf16* hin  = hfrag + (t & 1) * (BATCH * HCN);
    __bf16*       hout = hfrag + ((t + 1) & 1) * (BATCH * HCN);
    const size_t rowbase = (size_t)t * (BATCH * HCN);

    // warm the cache with this step's P rows while the WMMA chain runs
#pragma unroll
    for (int e = 0; e < 8; ++e)
      __builtin_prefetch(&P[rowbase + (size_t)(mt * 16 + mhalf + e) * HCN + hc], 0, 1);

    v8f acc = {};
#pragma unroll
    for (int kt = 0; kt < 16; ++kt) {
      v16bf a = *(const v16bf*)&hin[(kt * 2 + mt) * 512 + lane * 16];
      acc = __builtin_amdgcn_wmma_f32_16x16x32_bf16(false, a, false, wb[kt],
                                                    (short)0, acc, false, false);
    }

#pragma unroll
    for (int e = 0; e < 8; ++e) {
      int b = mt * 16 + mhalf + e;
      float z = acc[e] + P[rowbase + (size_t)b * HCN + hc] + bias;
      float hn = TANHF(z);
      ys[rowbase + (size_t)b * HCN + hc] = hn;
      __bf16 hb = (__bf16)hn;
      hout[fragbase + (mhalf + e + lane_hi) * 16 + e_a] = hb;   // next-step A frag
      if (ysb) ysb[rowbase + (size_t)b * HCN + hc] = hb;        // next-layer input
      if (t == Tn - 1) hs[b * HCN + hc] = hn;
    }

    // grid-wide step barrier (monotone counter; init kernel zeroes it)
    __threadfence();
    __syncthreads();
    if (tid == 0) {
      __hip_atomic_fetch_add(cnt, 1u, __ATOMIC_ACQ_REL, __HIP_MEMORY_SCOPE_AGENT);
      const unsigned target = (unsigned)SC_NWG * (unsigned)(t + 1);
      while (__hip_atomic_load(cnt, __ATOMIC_ACQUIRE, __HIP_MEMORY_SCOPE_AGENT) < target)
        __builtin_amdgcn_s_sleep(1);
    }
    __syncthreads();
  }
}

__global__ void init_kernel(unsigned* __restrict__ hfrag_u32, unsigned* __restrict__ cnt) {
  int i = blockIdx.x * 256 + threadIdx.x;
  if (i < (BATCH * HCN * 2) / 4) hfrag_u32[i] = 0u;   // h0 = 0 (bf16 zeros)
  if (i == 0) *cnt = 0u;
}

extern "C" void kernel_launch(void* const* d_in, const int* in_sizes, int n_in,
                              void* d_out, int out_size, void* d_ws, size_t ws_size,
                              hipStream_t stream) {
  const float* x  = (const float*)d_in[0];   // (T,B,IC)
  const float* Wi = (const float*)d_in[1];   // (L,HC,IC)
  const float* bi = (const float*)d_in[2];   // (L,HC)
  const float* Wh = (const float*)d_in[3];   // (L,HC,HC)
  const float* bh = (const float*)d_in[4];   // (L,HC)

  float* hs  = (float*)d_out;                 // (L,B,HC)
  float* inp = hs + 2 * BATCH * HCN;          // (T,B,HC) final-layer ys

  char* ws = (char*)d_ws;
  float*  P      = (float*)ws;                                        // 128 MB
  __bf16* Xbf    = (__bf16*)(ws + (size_t)MROWS * HCN * 4);           // 64 MB
  __bf16* WiBf   = (__bf16*)((char*)Xbf + (size_t)MROWS * HCN * 2);   // 512 KB
  __bf16* WhFrag = (__bf16*)((char*)WiBf + (size_t)HCN * ICN * 2);    // 512 KB
  __bf16* hfrag  = (__bf16*)((char*)WhFrag + (size_t)HCN * HCN * 2);  // 64 KB
  unsigned* cnt  = (unsigned*)((char*)hfrag + (size_t)2 * BATCH * HCN * 2);

  dim3 gx(MROWS / 128, HCN / 64);

  // ---- layer 0 ----
  cvt_kernel<<<(MROWS * HCN) / (256 * 8), 256, 0, stream>>>(x, Xbf, (long)MROWS * HCN);
  cvt_kernel<<<(HCN * ICN) / (256 * 8), 256, 0, stream>>>(Wi, WiBf, (long)HCN * ICN);
  whprep_kernel<<<SC_NWG, 256, 0, stream>>>(Wh, WhFrag);
  xproj_kernel<<<gx, 256, 0, stream>>>(Xbf, WiBf, bi, P);
  init_kernel<<<32, 256, 0, stream>>>((unsigned*)hfrag, cnt);
  // ys(f32) staged in d_out's inp region; bf16 mirror into Xbf for layer 1
  scan_kernel<<<SC_NWG, 256, 0, stream>>>(P, WhFrag, bh, inp, Xbf, hs, hfrag, cnt, T_STEPS);

  // ---- layer 1 ----
  cvt_kernel<<<(HCN * ICN) / (256 * 8), 256, 0, stream>>>(Wi + HCN * ICN, WiBf, (long)HCN * ICN);
  whprep_kernel<<<SC_NWG, 256, 0, stream>>>(Wh + HCN * HCN, WhFrag);
  xproj_kernel<<<gx, 256, 0, stream>>>(Xbf, WiBf, bi + HCN, P);
  init_kernel<<<32, 256, 0, stream>>>((unsigned*)hfrag, cnt);
  scan_kernel<<<SC_NWG, 256, 0, stream>>>(P, WhFrag, bh + HCN, inp, (__bf16*)nullptr,
                                          hs + BATCH * HCN, hfrag, cnt, T_STEPS);
}
